// GATv2_50087908606400
// MI455X (gfx1250) — compile-verified, matrix operand order checked by reading, not run on hardware
//
#include <hip/hip_runtime.h>
#include <cstdint>
#include <cstddef>

// Problem constants (match reference)
#define NN 50000
#define EE 800000
#define FF 32
#define HH 4
#define CC 32
#define HC 128
#define LL 3
#define NEG 0.2f

typedef __attribute__((ext_vector_type(16))) __bf16 v16bf;
typedef __attribute__((ext_vector_type(8)))  float v8f;

__device__ __forceinline__ float leaky(float v) { return v > 0.f ? v : NEG * v; }

__device__ __forceinline__ void atomicMaxF(float* addr, float val) {
    if (val >= 0.f) atomicMax((int*)addr, __float_as_int(val));
    else            atomicMin((unsigned int*)addr, __float_as_uint(val));
}

// ---------------------------------------------------------------------------
// Weight pre-pack: fp32 W[K, NC] -> bf16 fragments in the exact WMMA B-matrix
// register layout, so the GEMM loads one contiguous 32B chunk per lane.
// Fragment f = ks*NT + nt; within it, lane `l` (r=l&15 column, kb=(l>>4)*8)
// holds elems j<8 : K=ks*32+kb+j,  j>=8 : K=ks*32+kb+16+(j-8), col nt*16+r.
// ---------------------------------------------------------------------------
template <int K, int NC>
__global__ void pack_w_kernel(const float* __restrict__ W, __bf16* __restrict__ out)
{
    constexpr int KS = K / 32, NT = NC / 16;
    const int t = blockIdx.x * blockDim.x + threadIdx.x;
    if (t >= KS * NT * 32) return;
    const int lane = t & 31;
    const int nt   = (t >> 5) % NT;
    const int ks   = (t >> 5) / NT;
    const int r  = lane & 15;
    const int kb = (lane >> 4) * 8;
    __bf16* dstp = out + (size_t)t * 16;
    const float* wp = W + (size_t)(ks * 32) * NC + nt * 16 + r;
#pragma unroll
    for (int j = 0; j < 8; ++j) {
        dstp[j]     = (__bf16)wp[(size_t)(kb + j) * NC];
        dstp[j + 8] = (__bf16)wp[(size_t)(kb + 16 + j) * NC];
    }
}

// ---------------------------------------------------------------------------
// WMMA GEMM: Y[rows, NC] = X[rows, K] @ W[K, NC] + bias, optional epilogue
//   epilogue (EPI): y = leaky(y) * bsc[col] + bsh[col]
// One wave computes a 16-row x NC strip; 8 waves/block -> 128 rows/block.
// A: fp32 loaded as float4, converted to bf16 in regs (v_cvt_pk_bf16_f32).
// B: pre-packed bf16 fragments -> one 32B (2x b128) load per lane.
// f32 accumulate via v_wmma_f32_16x16x32_bf16.
// ---------------------------------------------------------------------------
template <int K, int NC, bool EPI>
__global__ void wmma_gemm_kernel(const float* __restrict__ X,
                                 const __bf16* __restrict__ Bp,
                                 const float* __restrict__ bias,
                                 const float* __restrict__ bsc,
                                 const float* __restrict__ bsh,
                                 float* __restrict__ Y, int rows)
{
    constexpr int KS = K / 32;   // k-steps
    constexpr int NT = NC / 16;  // column tiles
    const int lane = threadIdx.x & 31;
    const int wave = threadIdx.x >> 5;
    const int rowBase = blockIdx.x * 128 + wave * 16;
    if (rowBase >= rows) return;

    const int r  = lane & 15;
    const int kb = (lane >> 4) * 8;
    const int row = rowBase + r;

    v16bf a[KS];
#pragma unroll
    for (int ks = 0; ks < KS; ++ks) {
        const float* xp = X + (size_t)row * K + ks * 32;
        const float4 p0 = *(const float4*)(xp + kb);
        const float4 p1 = *(const float4*)(xp + kb + 4);
        const float4 q0 = *(const float4*)(xp + kb + 16);
        const float4 q1 = *(const float4*)(xp + kb + 20);
        a[ks][0]  = (__bf16)p0.x; a[ks][1]  = (__bf16)p0.y;
        a[ks][2]  = (__bf16)p0.z; a[ks][3]  = (__bf16)p0.w;
        a[ks][4]  = (__bf16)p1.x; a[ks][5]  = (__bf16)p1.y;
        a[ks][6]  = (__bf16)p1.z; a[ks][7]  = (__bf16)p1.w;
        a[ks][8]  = (__bf16)q0.x; a[ks][9]  = (__bf16)q0.y;
        a[ks][10] = (__bf16)q0.z; a[ks][11] = (__bf16)q0.w;
        a[ks][12] = (__bf16)q1.x; a[ks][13] = (__bf16)q1.y;
        a[ks][14] = (__bf16)q1.z; a[ks][15] = (__bf16)q1.w;
    }

#pragma unroll
    for (int nt = 0; nt < NT; ++nt) {
        const int ncol = nt * 16 + r;
        v8f c;
        const float bv = bias[ncol];
#pragma unroll
        for (int v = 0; v < 8; ++v) c[v] = bv;

#pragma unroll
        for (int ks = 0; ks < KS; ++ks) {
            const v16bf b =
                *(const v16bf*)(Bp + ((size_t)(ks * NT + nt) * 32 + lane) * 16);
            c = __builtin_amdgcn_wmma_f32_16x16x32_bf16(
                    false, a[ks], false, b, (short)0, c, false, false);
        }

        const int rowOut = rowBase + (lane >> 4) * 8;
        float sc = 1.f, sh = 0.f;
        if (EPI) { sc = bsc[ncol]; sh = bsh[ncol]; }
#pragma unroll
        for (int v = 0; v < 8; ++v) {
            float y = c[v];
            if (EPI) y = leaky(y) * sc + sh;
            Y[(size_t)(rowOut + v) * NC + ncol] = y;
        }
    }
}

// ---------------------------------------------------------------------------
// Init kernels
// ---------------------------------------------------------------------------
__global__ void init_md_kernel(float* __restrict__ m, float* __restrict__ denom, int n) {
    int t = blockIdx.x * blockDim.x + threadIdx.x;
    if (t < n) { m[t] = -INFINITY; denom[t] = 0.f; }
}

__global__ void init_out_kernel(float* __restrict__ out, const float* __restrict__ cb, int n) {
    int t = blockIdx.x * blockDim.x + threadIdx.x;
    if (t < n) out[t] = cb[t & (HC - 1)];
}

__global__ void init_gsum_kernel(float* __restrict__ g) {
    if (threadIdx.x < FF) g[threadIdx.x] = 0.f;
}

// ---------------------------------------------------------------------------
// Edge pass 1: logits[e,h] = sum_c leaky(xr[d]+xl[s]+|ea|*We)[h,c]*att[h,c]
// wave-per-edge, lane = channel -> coalesced 128B gathers (L2-resident tables)
// ---------------------------------------------------------------------------
__global__ void edge_logits_kernel(const float* __restrict__ xl, const float* __restrict__ xr,
                                   const int* __restrict__ src, const int* __restrict__ dst,
                                   const float* __restrict__ ea,
                                   const float* __restrict__ We, const float* __restrict__ att,
                                   float* __restrict__ logits, float* __restrict__ m)
{
    const int e = blockIdx.x * (blockDim.x >> 5) + (threadIdx.x >> 5);
    const int lane = threadIdx.x & 31;
    if (e >= EE) return;
    const int s = src[e], d = dst[e];
    const float w = fabsf(ea[e]);
#pragma unroll
    for (int h = 0; h < HH; ++h) {
        const int idx = h * CC + lane;
        float v = xr[(size_t)d * HC + idx] + xl[(size_t)s * HC + idx] + w * We[idx];
        float p = leaky(v) * att[idx];
#pragma unroll
        for (int off = 16; off; off >>= 1) p += __shfl_xor(p, off, 32);
        if (lane == 0) {
            logits[(size_t)e * HH + h] = p;
            atomicMaxF(&m[(size_t)d * HH + h], p);
        }
    }
}

// Edge pass 2: a = exp(logit - m[dst]); denom[dst] += a   (thread per edge*head)
__global__ void edge_exp_kernel(float* __restrict__ logits, const int* __restrict__ dst,
                                const float* __restrict__ m, float* __restrict__ denom)
{
    const int t = blockIdx.x * blockDim.x + threadIdx.x;
    if (t >= EE * HH) return;
    const int e = t >> 2, h = t & 3;
    const int d = dst[e];
    const float av = __expf(logits[t] - m[(size_t)d * HH + h]);
    logits[t] = av;  // overwrite in place with unnormalized alpha
    atomicAdd(&denom[(size_t)d * HH + h], av);
}

// Edge pass 3: out[dst] += xl[src] * (a / denom[dst])   (wave per edge)
__global__ void edge_scatter_kernel(const float* __restrict__ xl,
                                    const int* __restrict__ src, const int* __restrict__ dst,
                                    const float* __restrict__ a, const float* __restrict__ denom,
                                    float* __restrict__ out)
{
    const int e = blockIdx.x * (blockDim.x >> 5) + (threadIdx.x >> 5);
    const int lane = threadIdx.x & 31;
    if (e >= EE) return;
    const int s = src[e], d = dst[e];
#pragma unroll
    for (int h = 0; h < HH; ++h) {
        const float alpha = a[(size_t)e * HH + h] / denom[(size_t)d * HH + h];
        const int idx = h * CC + lane;
        atomicAdd(&out[(size_t)d * HC + idx], xl[(size_t)s * HC + idx] * alpha);
    }
}

// ---------------------------------------------------------------------------
// Mean pool: block-local reduction then 32 atomics per block
// ---------------------------------------------------------------------------
__global__ void pool_kernel(const float* __restrict__ h, float* __restrict__ gsum, int n)
{
    __shared__ float red[256];
    const int f = threadIdx.x & (FF - 1);
    const int rowsPerBlk = blockDim.x >> 5;  // 8 row-walkers (wave32)
    float acc = 0.f;
    for (int row = blockIdx.x * rowsPerBlk + (threadIdx.x >> 5); row < n;
         row += gridDim.x * rowsPerBlk)
        acc += h[(size_t)row * FF + f];
    red[threadIdx.x] = acc;
    __syncthreads();
    if (threadIdx.x < FF) {
        float s = 0.f;
        for (int i = threadIdx.x; i < 256; i += FF) s += red[i];
        atomicAdd(&gsum[threadIdx.x], s);
    }
}

// Head MLP: g -> 256 -> 32 -> 2 (single block, 256 threads)
__global__ void head_kernel(const float* __restrict__ gsum,
                            const float* __restrict__ W1, const float* __restrict__ b1,
                            const float* __restrict__ W2, const float* __restrict__ b2,
                            const float* __restrict__ W3, const float* __restrict__ b3,
                            float* __restrict__ outp, float invN)
{
    __shared__ float g[FF], t1[256], t2[FF];
    const int t = threadIdx.x;
    if (t < FF) g[t] = gsum[t] * invN;
    __syncthreads();
    {
        float acc = b1[t];
        for (int k = 0; k < FF; ++k) acc += g[k] * W1[k * 256 + t];
        t1[t] = leaky(acc);
    }
    __syncthreads();
    if (t < FF) {
        float acc = b2[t];
        for (int k = 0; k < 256; ++k) acc += t1[k] * W2[k * FF + t];
        t2[t] = leaky(acc);
    }
    __syncthreads();
    if (t < 2) {
        float acc = b3[t];
        for (int k = 0; k < FF; ++k) acc += t2[k] * W3[k * 2 + t];
        outp[t] = acc;
    }
}

// ---------------------------------------------------------------------------
extern "C" void kernel_launch(void* const* d_in, const int* in_sizes, int n_in,
                              void* d_out, int out_size, void* d_ws, size_t ws_size,
                              hipStream_t stream)
{
    const float* x    = (const float*)d_in[0];
    const int*   ei   = (const int*)d_in[1];      // [2,E]
    const float* ea   = (const float*)d_in[2];
    // d_in[3] = batch (all zeros, single graph) -- unused
    const float* Wl   = (const float*)d_in[4];    // [L,32,128]
    const float* bl   = (const float*)d_in[5];    // [L,128]
    const float* Wr   = (const float*)d_in[6];
    const float* br   = (const float*)d_in[7];
    const float* att  = (const float*)d_in[8];    // [L,4,32] = [L,128]
    const float* We   = (const float*)d_in[9];    // [L,1,128]
    const float* cb   = (const float*)d_in[10];   // [L,128]
    const float* linW = (const float*)d_in[11];   // [L,128,32]
    const float* linb = (const float*)d_in[12];   // [L,32]
    const float* bsc  = (const float*)d_in[13];   // [L,32]
    const float* bsh  = (const float*)d_in[14];   // [L,32]
    const float* W1   = (const float*)d_in[15];
    const float* b1   = (const float*)d_in[16];
    const float* W2   = (const float*)d_in[17];
    const float* b2   = (const float*)d_in[18];
    const float* W3   = (const float*)d_in[19];
    const float* b3   = (const float*)d_in[20];
    float* out = (float*)d_out;

    const int* src = ei;
    const int* dst = ei + EE;

    // Workspace layout (floats, then bf16 packed-weight region)
    float* ws     = (float*)d_ws;
    float* xl     = ws;                         // N*128
    float* xr     = xl     + (size_t)NN * HC;   // N*128
    float* outbuf = xr     + (size_t)NN * HC;   // N*128
    float* logits = outbuf + (size_t)NN * HC;   // E*4  (reused as alpha)
    float* mbuf   = logits + (size_t)EE * HH;   // N*4
    float* denom  = mbuf   + (size_t)NN * HH;   // N*4
    float* h0     = denom  + (size_t)NN * HH;   // N*32
    float* h1     = h0     + (size_t)NN * FF;   // N*32
    float* gsum   = h1     + (size_t)NN * FF;   // 32 (+pad to 64)
    // Packed bf16 weight fragments (32B-aligned): per layer Wl(4096) Wr(4096) linW(4096)
    __bf16* bpack = (__bf16*)(gsum + 64);
    const size_t FRAG = 4096;  // bf16 elems per packed matrix (KS*NT*32*16)

    const int gemmBlocks  = (NN + 127) / 128;   // 128 rows / block
    const int edgeBlocks  = (EE + 7) / 8;       // 8 waves (edges) / block
    const int ehBlocks    = (EE * HH + 255) / 256;
    const int mdBlocks    = (NN * HH + 255) / 256;
    const int outBlocks   = (NN * HC + 255) / 256;

    // One-time (per call) weight packing into WMMA fragment layout
    for (int l = 0; l < LL; ++l) {
        pack_w_kernel<32, 128><<<1, 256, 0, stream>>>(
            Wl + (size_t)l * FF * HC, bpack + (size_t)(l * 3 + 0) * FRAG);
        pack_w_kernel<32, 128><<<1, 256, 0, stream>>>(
            Wr + (size_t)l * FF * HC, bpack + (size_t)(l * 3 + 1) * FRAG);
        pack_w_kernel<128, 32><<<1, 256, 0, stream>>>(
            linW + (size_t)l * HC * FF, bpack + (size_t)(l * 3 + 2) * FRAG);
    }

    const float* hin = x;
    for (int l = 0; l < LL; ++l) {
        float* hout = (l & 1) ? h1 : h0;

        init_md_kernel <<<mdBlocks, 256, 0, stream>>>(mbuf, denom, NN * HH);
        init_out_kernel<<<outBlocks, 256, 0, stream>>>(outbuf, cb + (size_t)l * HC, NN * HC);

        // xl = hin @ Wl + bl ; xr = hin @ Wr + br   (single-K-pass WMMA, K=32)
        wmma_gemm_kernel<32, 128, false><<<gemmBlocks, 256, 0, stream>>>(
            hin, bpack + (size_t)(l * 3 + 0) * FRAG, bl + (size_t)l * HC,
            nullptr, nullptr, xl, NN);
        wmma_gemm_kernel<32, 128, false><<<gemmBlocks, 256, 0, stream>>>(
            hin, bpack + (size_t)(l * 3 + 1) * FRAG, br + (size_t)l * HC,
            nullptr, nullptr, xr, NN);

        edge_logits_kernel<<<edgeBlocks, 256, 0, stream>>>(
            xl, xr, src, dst, ea, We + (size_t)l * HC, att + (size_t)l * HC, logits, mbuf);
        edge_exp_kernel<<<ehBlocks, 256, 0, stream>>>(logits, dst, mbuf, denom);
        edge_scatter_kernel<<<edgeBlocks, 256, 0, stream>>>(xl, src, dst, logits, denom, outbuf);

        // hout = leaky(outbuf @ linW + linb) * bsc + bsh   (4 chained WMMAs, K=128)
        wmma_gemm_kernel<128, 32, true><<<gemmBlocks, 256, 0, stream>>>(
            outbuf, bpack + (size_t)(l * 3 + 2) * FRAG, linb + (size_t)l * FF,
            bsc + (size_t)l * FF, bsh + (size_t)l * FF, hout, NN);

        hin = hout;
    }

    // mean pool + head  (final h is in h0 after layer 2)
    init_gsum_kernel<<<1, 64, 0, stream>>>(gsum);
    pool_kernel<<<64, 256, 0, stream>>>(hin, gsum, NN);
    head_kernel<<<1, 256, 0, stream>>>(gsum, W1, b1, W2, b2, W3, b3, out, 1.0f / NN);
}
